// SparseAttention_8203387535661
// MI455X (gfx1250) — matrix operand order repaired and checked by reading, not compiled, not run on hardware
//
#include <hip/hip_runtime.h>
#include <hip/hip_bf16.h>
#include <math.h>

#define B_   2
#define H_   16
#define KVH_ 4
#define S_   4096
#define D_   64
#define BLK_ 64
#define W_   8
#define G_   (H_ / KVH_)   // 4
#define NQ_  (S_ / BLK_)   // 64
#define KVELEMS ((size_t)B_ * KVH_ * S_ * D_)   // 2,097,152

typedef __bf16 bf16;
typedef __attribute__((ext_vector_type(16))) __bf16 v16bf;
typedef __attribute__((ext_vector_type(8)))  __bf16 v8bf;
typedef __attribute__((ext_vector_type(8)))  float  v8f;
typedef __attribute__((ext_vector_type(4)))  unsigned int u32x4;
typedef __attribute__((ext_vector_type(8)))  int i32x8;
typedef __attribute__((ext_vector_type(4)))  int i32x4;

union V16U { v16bf v; v8bf h[2]; };

// ---------------------------------------------------------------------------
// Prep kernel: K -> bf16 (same layout); V -> bf16 transposed per 64x64 block
// so each KV tile becomes one contiguous 8KB blob the TDM can DMA into LDS.
// ---------------------------------------------------------------------------
__global__ __launch_bounds__(256)
void prep_kv_bf16_kernel(const float* __restrict__ k,
                         const float* __restrict__ v,
                         bf16* __restrict__ kb,
                         bf16* __restrict__ vt) {
    const size_t i = (size_t)blockIdx.x * 256 + threadIdx.x;
    if (i >= KVELEMS) return;
    kb[i] = (bf16)k[i];
    // v layout: [bk][blk][l][d] with bk = b*KVH+kvh ; write vt[bk][blk][d][l]
    const size_t d   = i & 63;
    const size_t l   = (i >> 6) & 63;
    const size_t blk = (i >> 12) & 63;
    const size_t bk  = i >> 18;
    vt[(((bk * 64 + blk) * 64) + d) * 64 + l] = (bf16)v[i];
}

// ---------------------------------------------------------------------------
// TDM: 1D copy of 4096 bf16 elements (8KB) from global to LDS.
// D# per CDNA5 ISA ch.8: group0 = {count/lds_addr/global_addr/type},
// group1 = {data_size=2B, tensor_dim0=tile_dim0=4096, stride=4096}.
// This toolchain uses the 6-arg builtin (clang-23 / therock-10.0 form).
// ---------------------------------------------------------------------------
__device__ __forceinline__ void tdm_load_tile_8k(unsigned lds_off,
                                                 const void* gptr) {
    const unsigned long long ga = (unsigned long long)(uintptr_t)gptr;
    u32x4 g0;
    g0[0] = 1u;                                       // count=1 (valid), user
    g0[1] = (unsigned)__builtin_amdgcn_readfirstlane((int)lds_off);
    g0[2] = (unsigned)__builtin_amdgcn_readfirstlane((int)(ga & 0xFFFFFFFFull));
    g0[3] = (unsigned)__builtin_amdgcn_readfirstlane(
                (int)(((ga >> 32) & 0x01FFFFFFull) | 0x80000000ull)); // type=2
    i32x8 g1;
    g1[0] = 0x00010000;            // workgroup_mask=0, data_size=1 (2 bytes)
    g1[1] = (int)(4096u << 16);    // tensor_dim0[15:0] in bits[63:48]
    g1[2] = 0;                     // tensor_dim0 hi=0, tensor_dim1 lo=0
    g1[3] = (int)(4096u << 16);    // tile_dim0=4096 in bits[127:112]
    g1[4] = 0;                     // tile_dim1=0 (1D), tile_dim2=0
    g1[5] = 4096;                  // tensor_dim0_stride low 32
    g1[6] = 0;
    g1[7] = 0;
    const i32x4 z4 = {0, 0, 0, 0};                   // groups 2/3 unused (<=2D)
    const i32x8 z8 = {0, 0, 0, 0, 0, 0, 0, 0};
    __builtin_amdgcn_tensor_load_to_lds(g0, g1, z4, z4, z8, 0);
}

// ---------------------------------------------------------------------------
// Attention kernel: one workgroup per (b, h, q-block); 4 waves of 32.
// ---------------------------------------------------------------------------
__global__ __launch_bounds__(128)
void sparse_attn_wmma_kernel(const float* __restrict__ q,
                             const bf16*  __restrict__ kb,
                             const bf16*  __restrict__ vt,
                             const float* __restrict__ alibi,
                             const int*   __restrict__ seg,
                             const int*   __restrict__ bidx,
                             float*       __restrict__ out) {
    const int bid = blockIdx.x;
    const int qb  = bid % NQ_;
    const int h   = (bid / NQ_) % H_;
    const int b   = bid / (NQ_ * H_);
    const int kvh = h / G_;

    const int tid  = threadIdx.x;
    const int lane = tid & 31;
    const int wave = tid >> 5;
    const int col  = lane & 15;            // N within a 16-wide tile / A-row M
    const int moff = (lane < 16) ? 0 : 8;  // C-layout row offset
    const int kbase = (lane < 16) ? 0 : 16; // A-layout K chunk base

    // Single flat LDS block so TDM byte offsets are well-defined.
    __shared__ __align__(128) unsigned char smem[8192 + 8192 + 8192 + 256 + 256];
    bf16* kLds    = (bf16*)(smem);             // K tile row-major [n][d], 8KB
    bf16* vtLds   = (bf16*)(smem + 8192);      // V tile transposed [d][l], 8KB
    bf16* pLds    = (bf16*)(smem + 16384);     // per-wave P staging, 8KB
    int*  segqLds = (int*)(smem + 24576);
    int*  segkLds = (int*)(smem + 24832);

    const float scale = 0.125f;            // 1/sqrt(64)
    const float slope = alibi[h];

    // ---- Q A-fragments (bf16), loaded once ----
    v16bf aq[2];
    {
        const float* qrow = q +
            (((size_t)b * H_ + h) * S_ + (size_t)qb * BLK_ + 16 * wave + col) * D_;
#pragma unroll
        for (int kt = 0; kt < 2; ++kt) {
            v16bf a;
#pragma unroll
            for (int e = 0; e < 8; ++e) {
                a[e]     = (bf16)qrow[32 * kt + kbase + e];
                a[8 + e] = (bf16)qrow[32 * kt + kbase + 8 + e];
            }
            aq[kt] = a;
        }
    }

    if (tid < 64)
        segqLds[tid] = seg[(size_t)b * S_ + (size_t)qb * BLK_ + tid];

    v8f oacc[4];
#pragma unroll
    for (int t = 0; t < 4; ++t)
#pragma unroll
        for (int r = 0; r < 8; ++r) oacc[t][r] = 0.0f;
    float run_max[8], run_l[8];
#pragma unroll
    for (int r = 0; r < 8; ++r) { run_max[r] = -1e30f; run_l[r] = 0.0f; }

    for (int w = 0; w < W_; ++w) {
        const int kblk = __builtin_amdgcn_readfirstlane(bidx[qb * W_ + w]);
        __syncthreads();   // previous iteration's tile reads complete
        if (kblk >= 0) {
            if (wave == 0) {
                const bf16* kg = kb +
                    (((size_t)b * KVH_ + kvh) * S_ + (size_t)kblk * BLK_) * D_;
                const bf16* vg = vt +
                    (((size_t)b * KVH_ + kvh) * NQ_ + kblk) * (size_t)(64 * 64);
                tdm_load_tile_8k((unsigned)(uintptr_t)kLds, kg);
                tdm_load_tile_8k((unsigned)(uintptr_t)vtLds, vg);
                __builtin_amdgcn_s_wait_tensorcnt(0);
            }
            if (tid < 64)
                segkLds[tid] = seg[(size_t)b * S_ + (size_t)kblk * BLK_ + tid];
        }
        __syncthreads();   // tiles (TDM-filled) visible to all waves
        if (kblk < 0) continue;

        // ---- S = Q @ K^T : 4 N-tiles x 2 K-steps ----
        v8f sacc[4];
#pragma unroll
        for (int t = 0; t < 4; ++t) {
            v8f c = {0.f, 0.f, 0.f, 0.f, 0.f, 0.f, 0.f, 0.f};
#pragma unroll
            for (int kt = 0; kt < 2; ++kt) {
                const bf16* kp =
                    &kLds[(16 * t + col) * 64 + 32 * kt + ((lane < 16) ? 0 : 16)];
                v16bf bk = *(const v16bf*)kp;
                c = __builtin_amdgcn_wmma_f32_16x16x32_bf16(
                        false, aq[kt], false, bk, (short)0, c, false, false);
            }
            sacc[t] = c;
        }

        // ---- scale + ALiBi + mask ----
#pragma unroll
        for (int t = 0; t < 4; ++t) {
            const int nl = 16 * t + col;
            const int ng = kblk * BLK_ + nl;
            const int sn = segkLds[nl];
#pragma unroll
            for (int r = 0; r < 8; ++r) {
                const int ml = 16 * wave + moff + r;
                const int mg = qb * BLK_ + ml;
                const int sq = segqLds[ml];
                const bool ok = (mg >= ng) && (sq == sn) && (sq >= 0);
                const float sv =
                    sacc[t][r] * scale - slope * fabsf((float)(mg - ng));
                sacc[t][r] = ok ? sv : -1e30f;
            }
        }

        // ---- online softmax (row reductions within 16-lane halves) ----
#pragma unroll
        for (int r = 0; r < 8; ++r) {
            float bm = fmaxf(fmaxf(sacc[0][r], sacc[1][r]),
                             fmaxf(sacc[2][r], sacc[3][r]));
            bm = fmaxf(bm, __shfl_xor(bm, 1, 32));
            bm = fmaxf(bm, __shfl_xor(bm, 2, 32));
            bm = fmaxf(bm, __shfl_xor(bm, 4, 32));
            bm = fmaxf(bm, __shfl_xor(bm, 8, 32));
            const float mnew = fmaxf(run_max[r], bm);
            const float corr = __expf(run_max[r] - mnew);
            run_max[r] = mnew;
            float psum = 0.0f;
#pragma unroll
            for (int t = 0; t < 4; ++t) {
                const float p = (sacc[t][r] > -1e29f)
                                    ? __expf(sacc[t][r] - mnew) : 0.0f;
                sacc[t][r] = p;
                psum += p;
            }
            psum += __shfl_xor(psum, 1, 32);
            psum += __shfl_xor(psum, 2, 32);
            psum += __shfl_xor(psum, 4, 32);
            psum += __shfl_xor(psum, 8, 32);
            run_l[r] = run_l[r] * corr + psum;
#pragma unroll
            for (int t = 0; t < 4; ++t) oacc[t][r] *= corr;
        }

        // ---- P -> bf16 via per-wave LDS staging (same-wave DS ops in order) --
        bf16* pw = &pLds[wave * 16 * 64];
#pragma unroll
        for (int t = 0; t < 4; ++t)
#pragma unroll
            for (int r = 0; r < 8; ++r)
                pw[(moff + r) * 64 + 16 * t + col] = (bf16)sacc[t][r];

        v16bf pa[2];
        {
            const bf16* prow = pw + col * 64;
#pragma unroll
            for (int kt = 0; kt < 2; ++kt) {
                V16U u;
                u.h[0] = *(const v8bf*)(prow + 32 * kt + kbase);
                u.h[1] = *(const v8bf*)(prow + 32 * kt + kbase + 8);
                pa[kt] = u.v;
            }
        }

        // ---- O += P @ V ----
#pragma unroll
        for (int t = 0; t < 4; ++t) {
#pragma unroll
            for (int kt = 0; kt < 2; ++kt) {
                const bf16* vp =
                    &vtLds[(16 * t + col) * 64 + 32 * kt + ((lane < 16) ? 0 : 16)];
                v16bf bv = *(const v16bf*)vp;
                oacc[t] = __builtin_amdgcn_wmma_f32_16x16x32_bf16(
                              false, pa[kt], false, bv, (short)0, oacc[t],
                              false, false);
            }
        }
    }

    // ---- normalize and write out ----
    float* op = out + (((size_t)b * H_ + h) * S_ + (size_t)qb * BLK_ + 16 * wave) * D_;
#pragma unroll
    for (int t = 0; t < 4; ++t)
#pragma unroll
        for (int r = 0; r < 8; ++r) {
            const float l = run_l[r];
            const float val = (l > 0.0f) ? (oacc[t][r] / l) : 0.0f;
            op[(moff + r) * 64 + 16 * t + col] = val;
        }
}

extern "C" void kernel_launch(void* const* d_in, const int* in_sizes, int n_in,
                              void* d_out, int out_size, void* d_ws, size_t ws_size,
                              hipStream_t stream) {
    const float* q     = (const float*)d_in[0];
    const float* k     = (const float*)d_in[1];
    const float* v     = (const float*)d_in[2];
    const float* alibi = (const float*)d_in[3];
    const int*   seg   = (const int*)d_in[4];
    const int*   bidx  = (const int*)d_in[5];
    float* out = (float*)d_out;

    bf16* kb = (bf16*)d_ws;              // 4 MiB
    bf16* vt = kb + KVELEMS;             // 4 MiB

    {
        const int n = (int)KVELEMS;
        prep_kv_bf16_kernel<<<(n + 255) / 256, 256, 0, stream>>>(k, v, kb, vt);
    }
    {
        dim3 grid(B_ * H_ * NQ_);        // 2048 workgroups
        dim3 block(128);                 // 4 waves of 32
        sparse_attn_wmma_kernel<<<grid, block, 0, stream>>>(q, kb, vt, alibi,
                                                            seg, bidx, out);
    }
}
